// OptimizedGraphTransGeoGAT_78546361909452
// MI455X (gfx1250) — matrix operand order repaired and checked by reading, not compiled, not run on hardware
//
#include <hip/hip_runtime.h>
#include <hip/hip_bf16.h>

// ---------------------------------------------------------------------------
// GAT (3 layers, 4 heads x 64ch) on MI455X / gfx1250.
// Dense GEMMs: v_wmma_f32_16x16x32_f16 with double-buffered LDS tiles filled
// by global_load_async_to_lds_b128 (ASYNCcnt). Graph scatter: L2-resident
// f32 atomics (node features = 51MB << 192MB L2).
// ---------------------------------------------------------------------------

typedef __attribute__((ext_vector_type(16))) _Float16 v16h;
typedef __attribute__((ext_vector_type(8)))  _Float16 v8h;
typedef __attribute__((ext_vector_type(8)))  float    v8f;

#define HEADS 4
#define CH 64
#define HID 256
#define NGRAPH 64
#define BN_EPS 1e-5f

// ---------------- CDNA5 async global->LDS copy helpers ----------------------
__device__ __forceinline__ void async_ld_b128(const void* lds_dst, const void* gaddr) {
#if defined(__gfx1250__)
  // LDS byte address == low 32 bits of the flat pointer (LDS aperture mapping)
  unsigned ldsoff = (unsigned)(unsigned long long)(size_t)lds_dst;
  unsigned long long ga = (unsigned long long)(size_t)gaddr;
  asm volatile("global_load_async_to_lds_b128 %0, %1, off"
               :: "v"(ldsoff), "v"(ga) : "memory");
#else
  // host-pass / non-gfx1250 fallback: plain copy
  *(v8h*)lds_dst = *(const v8h*)gaddr;
#endif
}
__device__ __forceinline__ void wait_async0() {
#if defined(__gfx1250__)
  asm volatile("s_wait_asynccnt 0x0" ::: "memory");
#endif
}

// ---------------- f32 -> f16 conversion ------------------------------------
__global__ void cvt_f32_f16(const float* __restrict__ in, _Float16* __restrict__ out, size_t n) {
  size_t stride = (size_t)gridDim.x * blockDim.x;
  for (size_t i = (size_t)blockIdx.x * blockDim.x + threadIdx.x; i < n; i += stride)
    out[i] = (_Float16)in[i];
}

// W[K][HID] (f32) -> Wt[HID][K] (f16), done once per layer so the GEMM can
// stage B tiles with contiguous 16B async copies (no in-LDS transpose).
__global__ void cvt_transpose_w(const float* __restrict__ W, _Float16* __restrict__ Wt, int K) {
  int i = blockIdx.x * blockDim.x + threadIdx.x;   // over K*HID
  if (i >= K * HID) return;
  int k = i / HID, n = i % HID;
  Wt[n * K + k] = (_Float16)W[i];
}

// ---------------- WMMA GEMM: C[N,256] = A[N,K] @ Wt^T -----------------------
// Block: 256 threads = 8 waves. Tile: 128(M) x 64(N), K-step 32, double-buffered.
// Wave w owns rows w*16..w*16+15 and sweeps 4 N-subtiles of 16.
#define TM 128
#define TN 64
#define TK 32

__global__ __launch_bounds__(256)
void gemm_f16_wmma(const _Float16* __restrict__ A,     // [N,K] row-major
                   const _Float16* __restrict__ Wt,    // [HID,K] row-major (pre-transposed)
                   float* __restrict__ C, int N, int K) {
  __shared__ _Float16 sA[2][TM * TK];                  // [m][k]
  __shared__ _Float16 sBt[2][TN * TK];                 // [n][k]

  const int tid  = threadIdx.x;
  const int lane = tid & 31;
  const int wave = tid >> 5;
  const int mBase = blockIdx.x * TM;
  const int nBase = blockIdx.y * TN;

  // per-thread staging coordinates (3 async b128 copies per thread per tile)
  // A: 512 chunks of 8 halfs -> 2 per thread
  int ar0 = tid >> 1;                  // wrong split? no: chunk c = tid, tid+256
  (void)ar0;
  const int aR0 = (tid)       >> 2, aC0 = ((tid)       & 3) * 8;
  const int aR1 = (tid + 256) >> 2, aC1 = ((tid + 256) & 3) * 8;
  int gA0 = mBase + aR0; if (gA0 >= N) gA0 = N - 1;
  int gA1 = mBase + aR1; if (gA1 >= N) gA1 = N - 1;
  // B: 256 chunks of 8 halfs -> 1 per thread (64 rows x 4 chunks)
  const int bR = tid >> 2, bC = (tid & 3) * 8;

  const int kTiles = K / TK;

  // ---- prologue: stage tile 0 into buffer 0
  {
    const int k0 = 0;
    async_ld_b128(&sA[0][aR0 * TK + aC0], A + (size_t)gA0 * K + k0 + aC0);
    async_ld_b128(&sA[0][aR1 * TK + aC1], A + (size_t)gA1 * K + k0 + aC1);
    async_ld_b128(&sBt[0][bR * TK + bC], Wt + (size_t)(nBase + bR) * K + k0 + bC);
  }
  wait_async0();
  __syncthreads();

  v8f acc[4] = {};

  for (int kt = 0; kt < kTiles; ++kt) {
    const int cur = kt & 1;
    // ---- stage next tile into the other buffer while we compute
    if (kt + 1 < kTiles) {
      const int k0 = (kt + 1) * TK;
      const int nb = cur ^ 1;
      async_ld_b128(&sA[nb][aR0 * TK + aC0], A + (size_t)gA0 * K + k0 + aC0);
      async_ld_b128(&sA[nb][aR1 * TK + aC1], A + (size_t)gA1 * K + k0 + aC1);
      async_ld_b128(&sBt[nb][bR * TK + bC], Wt + (size_t)(nBase + bR) * K + k0 + bC);
    }

    // --- A fragment (16x32 f16): lane<16 holds K 0..7 & 16..23, lane>=16 holds
    // K 8..15 & 24..31 (ISA 16-bit A layout) -> two contiguous 16B LDS loads.
    const int m  = wave * 16 + (lane & 15);
    const int kb = (lane >> 4) * 8;
    v8h alo = *(const v8h*)(&sA[cur][m * TK + kb]);
    v8h ahi = *(const v8h*)(&sA[cur][m * TK + kb + 16]);
    v16h af = __builtin_shufflevector(alo, ahi, 0,1,2,3,4,5,6,7,8,9,10,11,12,13,14,15);

    // --- 4 B fragments (32x16 f16): lane holds column n = lane%16,
    // lane<16 holds K 0..7 & 16..23, lane>=16 holds K 8..15 & 24..31 (matches A).
    #pragma unroll
    for (int t = 0; t < 4; ++t) {
      const int n    = t * 16 + (lane & 15);
      const int koff = (lane >> 4) * 8;
      v8h blo = *(const v8h*)(&sBt[cur][n * TK + koff]);
      v8h bhi = *(const v8h*)(&sBt[cur][n * TK + koff + 16]);
      v16h bf = __builtin_shufflevector(blo, bhi, 0,1,2,3,4,5,6,7,8,9,10,11,12,13,14,15);
      acc[t] = __builtin_amdgcn_wmma_f32_16x16x32_f16(
          false, af, false, bf, (short)0, acc[t], false, false);
    }

    wait_async0();       // our async copies for next tile have landed
    __syncthreads();     // everyone's copies landed + everyone done reading cur
  }

  // --- store: C/D layout: VGPR v -> row = v + 8*(lane>=16), col = lane%16
  #pragma unroll
  for (int t = 0; t < 4; ++t) {
    #pragma unroll
    for (int v = 0; v < 8; ++v) {
      int gr = mBase + wave * 16 + v + 8 * (lane >> 4);
      int gc = nBase + t * 16 + (lane & 15);
      if (gr < N) C[(size_t)gr * HID + gc] = acc[t][v];
    }
  }
}

// ---------------- BatchNorm (training-mode, biased var) ---------------------
__global__ void bn_stats(const float* __restrict__ x, float* __restrict__ sum,
                         float* __restrict__ sumsq, int N, int C) {
  int col = threadIdx.x;                    // blockDim.x == C
  float s = 0.f, q = 0.f;
  for (int r = blockIdx.x; r < N; r += gridDim.x) {
    float v = x[(size_t)r * C + col];
    s += v; q += v * v;
  }
  atomicAdd(&sum[col], s);
  atomicAdd(&sumsq[col], q);
}

// mode 0: dst = norm(x); mode 1: dst += norm(x)  (residual)
__global__ void bn_apply(const float* __restrict__ x, float* __restrict__ dst,
                         const float* __restrict__ sum, const float* __restrict__ sumsq,
                         const float* __restrict__ gamma, const float* __restrict__ beta,
                         int N, int C, int mode) {
  size_t total = (size_t)N * C;
  size_t stride = (size_t)gridDim.x * blockDim.x;
  for (size_t i = (size_t)blockIdx.x * blockDim.x + threadIdx.x; i < total; i += stride) {
    int c = (int)(i % C);
    float mu  = sum[c] / (float)N;
    float var = sumsq[c] / (float)N - mu * mu;
    float nv  = (x[i] - mu) * __frsqrt_rn(var + BN_EPS) * gamma[c] + beta[c];
    if (mode) dst[i] += nv; else dst[i] = nv;
  }
}

// ---------------- attention: per-node coefficients --------------------------
__global__ void att_coef(const float* __restrict__ h, const float* __restrict__ a_src,
                         const float* __restrict__ a_dst, float* __restrict__ al_s,
                         float* __restrict__ al_d, int N) {
  int i = blockIdx.x * blockDim.x + threadIdx.x;   // over N*HEADS
  if (i >= N * HEADS) return;
  int n = i >> 2, hh = i & 3;
  const float* hp = h + (size_t)n * HID + hh * CH;
  const float* as = a_src + hh * CH;
  const float* ad = a_dst + hh * CH;
  float s = 0.f, d = 0.f;
  #pragma unroll 8
  for (int c = 0; c < CH; ++c) { float v = hp[c]; s += v * as[c]; d += v * ad[c]; }
  al_s[i] = s; al_d[i] = d;
}

// ordered-uint mapping so atomicMax(u32) == float max
__device__ inline unsigned f2ord(float f) {
  unsigned b = __float_as_uint(f);
  return (b & 0x80000000u) ? ~b : (b | 0x80000000u);
}
__device__ inline float ord2f(unsigned u) {
  return (u & 0x80000000u) ? __uint_as_float(u & 0x7FFFFFFFu) : __uint_as_float(~u);
}

// pass A: leaky_relu logits + segment max (self-loops are edges e >= E)
__global__ void edge_logits_max(const int* __restrict__ src, const int* __restrict__ dst,
                                const float* __restrict__ al_s, const float* __restrict__ al_d,
                                float* __restrict__ ebuf, unsigned* __restrict__ mbits,
                                int E, int N) {
  int Etot = E + N;
  int i = blockIdx.x * blockDim.x + threadIdx.x;   // over Etot*HEADS
  if (i >= Etot * HEADS) return;
  int e = i >> 2, hh = i & 3;
  int s, d;
  if (e < E) { s = src[e]; d = dst[e]; } else { s = d = e - E; }
  float v = al_s[s * HEADS + hh] + al_d[d * HEADS + hh];
  v = v > 0.f ? v : 0.2f * v;
  ebuf[i] = v;
  atomicMax(&mbits[d * HEADS + hh], f2ord(v));
}

// pass B: exp(e - max) + segment sum
__global__ void edge_exp_sum(const int* __restrict__ src, const int* __restrict__ dst,
                             float* __restrict__ ebuf, const unsigned* __restrict__ mbits,
                             float* __restrict__ denom, int E, int N) {
  int Etot = E + N;
  int i = blockIdx.x * blockDim.x + threadIdx.x;
  if (i >= Etot * HEADS) return;
  int e = i >> 2, hh = i & 3;
  int d = (e < E) ? dst[e] : (e - E);
  float m  = ord2f(mbits[d * HEADS + hh]);
  float ex = __expf(ebuf[i] - m);
  ebuf[i] = ex;
  atomicAdd(&denom[d * HEADS + hh], ex);
}

// pass C: one wave per edge; lane covers 8 channels; scatter-add weighted msg
__global__ __launch_bounds__(256)
void edge_scatter(const int* __restrict__ src, const int* __restrict__ dst,
                  const float* __restrict__ h, const float* __restrict__ ebuf,
                  const float* __restrict__ denom, float* __restrict__ agg,
                  int E, int N) {
  int Etot = E + N;
  int e = blockIdx.x * 8 + (threadIdx.x >> 5);
  if (e >= Etot) return;
  int lane = threadIdx.x & 31;
  int s, d;
  if (e < E) { s = src[e]; d = dst[e]; } else { s = d = e - E; }
  int hh = lane >> 3;                                 // 8 lanes per head
  float alpha = ebuf[e * HEADS + hh] / denom[d * HEADS + hh];
  const float4* hp = (const float4*)(h + (size_t)s * HID + lane * 8); // L2 gather
  float4 a0 = hp[0], a1 = hp[1];
  float* op = agg + (size_t)d * HID + lane * 8;
  atomicAdd(&op[0], a0.x * alpha); atomicAdd(&op[1], a0.y * alpha);
  atomicAdd(&op[2], a0.z * alpha); atomicAdd(&op[3], a0.w * alpha);
  atomicAdd(&op[4], a1.x * alpha); atomicAdd(&op[5], a1.y * alpha);
  atomicAdd(&op[6], a1.z * alpha); atomicAdd(&op[7], a1.w * alpha);
}

// bias + relu, in place
__global__ void bias_relu(float* __restrict__ x, const float* __restrict__ b, int N) {
  size_t total = (size_t)N * HID;
  size_t stride = (size_t)gridDim.x * blockDim.x;
  for (size_t i = (size_t)blockIdx.x * blockDim.x + threadIdx.x; i < total; i += stride) {
    float v = x[i] + b[i % HID];
    x[i] = v > 0.f ? v : 0.f;
  }
}

// ---------------- global mean pool + output linear --------------------------
__global__ void pool_add(const float* __restrict__ x, const int* __restrict__ batch,
                         float* __restrict__ psum, float* __restrict__ pcnt, int N) {
  size_t i = (size_t)blockIdx.x * blockDim.x + threadIdx.x;   // over N*HID
  if (i >= (size_t)N * HID) return;
  int n = (int)(i >> 8), c = (int)(i & 255);
  int g = batch[n];
  atomicAdd(&psum[(size_t)g * HID + c], x[i]);
  if (c == 0) atomicAdd(&pcnt[g], 1.0f);
}

__global__ void final_linear(const float* __restrict__ psum, const float* __restrict__ pcnt,
                             const float* __restrict__ Wout, const float* __restrict__ bout,
                             float* __restrict__ out) {
  int i = blockIdx.x * blockDim.x + threadIdx.x;   // NGRAPH*2
  if (i >= NGRAPH * 2) return;
  int g = i >> 1, j = i & 1;
  float inv = 1.0f / fmaxf(pcnt[g], 1.0f);
  float acc = bout[j];
  #pragma unroll 8
  for (int c = 0; c < HID; ++c) acc += psum[(size_t)g * HID + c] * inv * Wout[c * 2 + j];
  out[i] = acc;
}

// ---------------------------------------------------------------------------
static inline size_t alignup(size_t v) { return (v + 255) & ~(size_t)255; }

extern "C" void kernel_launch(void* const* d_in, const int* in_sizes, int n_in,
                              void* d_out, int out_size, void* d_ws, size_t ws_size,
                              hipStream_t stream) {
  const int FIN = 128;
  const int N = in_sizes[0] / FIN;
  const int E = in_sizes[1] / 2;
  const int Etot = E + N;

  const float* x     = (const float*)d_in[0];
  const int*   edge  = (const int*)d_in[1];
  const int*   batch = (const int*)d_in[2];
  const float* bn_g  = (const float*)d_in[3];
  const float* bn_b  = (const float*)d_in[4];
  const int*   srcE  = edge;
  const int*   dstE  = edge + E;
  float* out = (float*)d_out;

  // workspace carving
  char* p = (char*)d_ws; size_t off = 0;
  auto take = [&](size_t bytes) { char* r = p + off; off = alignup(off + bytes); return (void*)r; };
  float*     xprev = (float*)take((size_t)N * HID * 4);
  float*     hbuf  = (float*)take((size_t)N * HID * 4);
  float*     agg   = (float*)take((size_t)N * HID * 4);
  float*     bnin  = (float*)take((size_t)N * FIN * 4);
  _Float16*  xf16  = (_Float16*)take((size_t)N * HID * 2);
  _Float16*  wt16  = (_Float16*)take((size_t)HID * HID * 2);
  float*     al_s  = (float*)take((size_t)N * HEADS * 4);
  float*     al_d  = (float*)take((size_t)N * HEADS * 4);
  unsigned*  mbits = (unsigned*)take((size_t)N * HEADS * 4);
  float*     denom = (float*)take((size_t)N * HEADS * 4);
  float*     ebuf  = (float*)take((size_t)Etot * HEADS * 4);
  float*     ssum  = (float*)take(HID * 4);
  float*     ssq   = (float*)take(HID * 4);
  float*     psum  = (float*)take((size_t)NGRAPH * HID * 4);
  float*     pcnt  = (float*)take(NGRAPH * 4);
  (void)ws_size;

  const int T = 256;
  const int gridEl = 2048;   // grid-stride elementwise kernels

  // ---- input batchnorm ----
  hipMemsetAsync(ssum, 0, HID * 4, stream);
  hipMemsetAsync(ssq,  0, HID * 4, stream);
  bn_stats<<<240, FIN, 0, stream>>>(x, ssum, ssq, N, FIN);
  bn_apply<<<gridEl, T, 0, stream>>>(x, bnin, ssum, ssq, bn_g, bn_b, N, FIN, 0);

  for (int l = 0; l < 3; ++l) {
    const float* W  = (const float*)d_in[5 + 6 * l + 0];
    const float* as = (const float*)d_in[5 + 6 * l + 1];
    const float* ad = (const float*)d_in[5 + 6 * l + 2];
    const float* bi = (const float*)d_in[5 + 6 * l + 3];
    const float* ga = (const float*)d_in[5 + 6 * l + 4];
    const float* be = (const float*)d_in[5 + 6 * l + 5];
    const float* in_f32 = (l == 0) ? bnin : xprev;
    const int K = (l == 0) ? FIN : HID;

    // convert inputs to f16; transpose+convert weights
    cvt_f32_f16<<<gridEl, T, 0, stream>>>(in_f32, xf16, (size_t)N * K);
    cvt_transpose_w<<<(K * HID + T - 1) / T, T, 0, stream>>>(W, wt16, K);

    // H = X @ W   (v_wmma_f32_16x16x32_f16, async double-buffered LDS)
    dim3 gg((N + TM - 1) / TM, HID / TN);
    gemm_f16_wmma<<<gg, T, 0, stream>>>(xf16, wt16, hbuf, N, K);

    // attention coefficients per (node, head)
    att_coef<<<(N * HEADS + T - 1) / T, T, 0, stream>>>(hbuf, as, ad, al_s, al_d, N);

    // softmax over incoming edges
    hipMemsetAsync(mbits, 0, (size_t)N * HEADS * 4, stream);   // ordered-key for -inf
    hipMemsetAsync(denom, 0, (size_t)N * HEADS * 4, stream);
    int nEH = Etot * HEADS;
    edge_logits_max<<<(nEH + T - 1) / T, T, 0, stream>>>(srcE, dstE, al_s, al_d, ebuf, mbits, E, N);
    edge_exp_sum<<<(nEH + T - 1) / T, T, 0, stream>>>(srcE, dstE, ebuf, mbits, denom, E, N);

    // weighted message aggregation (wave per edge)
    hipMemsetAsync(agg, 0, (size_t)N * HID * 4, stream);
    edge_scatter<<<(Etot + 7) / 8, T, 0, stream>>>(srcE, dstE, hbuf, ebuf, denom, agg, E, N);

    // bias + relu, then BN (+ residual for layers 1,2)
    bias_relu<<<gridEl, T, 0, stream>>>(agg, bi, N);
    hipMemsetAsync(ssum, 0, HID * 4, stream);
    hipMemsetAsync(ssq,  0, HID * 4, stream);
    bn_stats<<<240, HID, 0, stream>>>(agg, ssum, ssq, N, HID);
    bn_apply<<<gridEl, T, 0, stream>>>(agg, xprev, ssum, ssq, ga, be, N, HID, (l == 0) ? 0 : 1);
  }

  // ---- global mean pool + output projection ----
  hipMemsetAsync(psum, 0, (size_t)NGRAPH * HID * 4, stream);
  hipMemsetAsync(pcnt, 0, NGRAPH * 4, stream);
  size_t nNC = (size_t)N * HID;
  pool_add<<<(unsigned)((nNC + T - 1) / T), T, 0, stream>>>(xprev, batch, psum, pcnt, N);
  final_linear<<<1, 128, 0, stream>>>(psum, pcnt, (const float*)d_in[23], (const float*)d_in[24], out);
  (void)out_size; (void)n_in;
}
